// Circorr1d_1692217114965
// MI455X (gfx1250) — matrix operand order repaired and checked by reading, not compiled, not run on hardware
//
#include <hip/hip_runtime.h>

// CDNA5 / gfx1250 wave32 WMMA circular-correlation kernel.
// y[b,o,n] = sum_c sum_m w[o,c,m] * x[b,c,(n+m)&4095]  (+bias[o])
// bf16x3 split-GEMM (hi/lo decomposition, f32 accumulate).
// Fast path: prep kernels split x/w to bf16 in workspace (position-major),
// main kernel stages tiles with global_load_async_to_lds_b128 (ASYNCcnt).
// Fallback path (small ws): round-1 all-in-one kernel.

typedef __attribute__((ext_vector_type(16))) __bf16 v16bf;
typedef __attribute__((ext_vector_type(8)))  float  v8f;

#define B_SZ   32
#define C_IN   128
#define C_OUT  128
#define KSIZE  9
#define NLEN   4096
#define NT     64            // output positions per block
#define XCOLS  (NT + 8)      // 72 staged columns (extra tail for taps)
#define XSTR   (C_IN + 8)    // 136: channel stride within a column (16B-aligned reads)
#define WSTR   40            // 32 chunk channels + pad (80B rows, 16B aligned)

#define X_ELEMS ((size_t)B_SZ * NLEN * C_IN)        // 16,777,216 bf16 per array
#define W_ELEMS ((size_t)KSIZE * C_OUT * C_IN)      // 147,456 bf16 per array
#define WS_NEED ((2 * X_ELEMS + 2 * W_ELEMS) * 2)   // bytes

__device__ __forceinline__ unsigned short bf16_rne(float f) {
  unsigned u = __float_as_uint(f);
  u += 0x7fffu + ((u >> 16) & 1u);          // round-to-nearest-even
  return (unsigned short)(u >> 16);
}
__device__ __forceinline__ float bf16_tof(unsigned short h) {
  return __uint_as_float(((unsigned)h) << 16);
}

union Frag {
  v16bf v;
  uint4 q[2];
};

// ---- CDNA5 async global->LDS copy (16B per lane), ASYNCcnt tracked ----
__device__ __forceinline__ unsigned lds_off(const void* p) {
  return (unsigned)(uintptr_t)(__attribute__((address_space(3))) const void*)p;
}
__device__ __forceinline__ void async_cp16(const void* gsrc, void* ldst) {
  unsigned l = lds_off(ldst);
  unsigned long long g = (unsigned long long)(uintptr_t)gsrc;
  asm volatile("global_load_async_to_lds_b128 %0, %1, off" :: "v"(l), "v"(g) : "memory");
}
__device__ __forceinline__ void async_wait0() {
  asm volatile("s_wait_asynccnt 0x0" ::: "memory");
}

// =====================================================================
// Prep 1: split x (f32 [b][c][n]) -> bf16 hi/lo, transposed to [b][n][c]
// so each position's 128-channel vector is one contiguous 256B run.
// =====================================================================
__global__ __launch_bounds__(256) void split_x(
    const float* __restrict__ x,
    unsigned short* __restrict__ xhi, unsigned short* __restrict__ xlo) {
  __shared__ unsigned short Th[64 * XSTR];
  __shared__ unsigned short Tl[64 * XSTR];
  const int tid = threadIdx.x;
  const int n0  = blockIdx.x * 64;
  const int b   = blockIdx.y;
  const float* xb = x + (size_t)b * C_IN * NLEN;
  for (int t = tid; t < C_IN * 64; t += 256) {
    int c = t >> 6, j = t & 63;                 // coalesced along n
    float v = xb[(size_t)c * NLEN + n0 + j];
    unsigned short h = bf16_rne(v);
    Th[j * XSTR + c] = h;
    Tl[j * XSTR + c] = bf16_rne(v - bf16_tof(h));
  }
  __syncthreads();
  size_t obase = ((size_t)(b * NLEN + n0)) << 7;   // *C_IN
  for (int t = tid; t < C_IN * 64; t += 256) {
    int j = t >> 7, c = t & 127;                // coalesced along c
    xhi[obase + ((size_t)j << 7) + c] = Th[j * XSTR + c];
    xlo[obase + ((size_t)j << 7) + c] = Tl[j * XSTR + c];
  }
}

// =====================================================================
// Prep 2: split weights (f32 [o][c][m]) -> bf16 hi/lo as [m][o][c]
// (per-tap 128x128 row-major matrices, contiguous 256B rows).
// =====================================================================
__global__ __launch_bounds__(256) void split_w(
    const float* __restrict__ w,
    unsigned short* __restrict__ whi, unsigned short* __restrict__ wlo) {
  int g = blockIdx.x * 256 + threadIdx.x;
  if (g >= (int)W_ELEMS) return;
  int c = g & 127, o = (g >> 7) & 127, m = g >> 14;   // g = (m*128+o)*128+c
  float v = w[o * (C_IN * KSIZE) + c * KSIZE + m];
  unsigned short h = bf16_rne(v);
  whi[g] = h;
  wlo[g] = bf16_rne(v - bf16_tof(h));
}

// =====================================================================
// Main GEMM kernel — async-to-LDS staging variant.
// =====================================================================
__global__ __launch_bounds__(256) void circorr_wmma_async(
    const unsigned short* __restrict__ xhi, const unsigned short* __restrict__ xlo,
    const unsigned short* __restrict__ whi, const unsigned short* __restrict__ wlo,
    const float* __restrict__ bias, float* __restrict__ out) {
  __shared__ __align__(16) unsigned short XsHi[XCOLS * XSTR];
  __shared__ __align__(16) unsigned short XsLo[XCOLS * XSTR];
  __shared__ __align__(16) unsigned short WsHi[C_OUT * WSTR];
  __shared__ __align__(16) unsigned short WsLo[C_OUT * WSTR];
  __shared__ float biasS[C_OUT];

  const int tid  = threadIdx.x;
  const int lane = tid & 31;
  const int wave = tid >> 5;
  const int wr   = wave >> 1;      // 0..3 : 32-row group
  const int wc   = wave & 1;       // 0..1 : 32-col group
  const int l15  = lane & 15;
  const int half = lane >> 4;

  const int nb     = blockIdx.x;   // 0..63 position tile
  const int b      = blockIdx.y;   // 0..31 batch
  const int n_base = nb * NT;

  // ---- stage X tile: 72 columns x 128ch, each column = 16 lanes x b128 per array ----
  for (int t = tid; t < XCOLS * 16; t += 256) {
    int col = t >> 4;                          // 0..71
    int seg = t & 15;                          // 16B segment within column
    int cs  = (n_base + col) & (NLEN - 1);     // circular wrap (column granularity)
    size_t gi = (((size_t)(b * NLEN + cs)) << 7) + (seg << 3);
    async_cp16(xhi + gi, &XsHi[col * XSTR + (seg << 3)]);
    async_cp16(xlo + gi, &XsLo[col * XSTR + (seg << 3)]);
  }
  if (tid < C_OUT) biasS[tid] = bias[tid];
  async_wait0();
  __syncthreads();

  v8f acc[2][2] = {};

  for (int m = 0; m < KSIZE; ++m) {
    for (int cc = 0; cc < C_IN / 32; ++cc) {
      __syncthreads();   // previous chunk's Ws reads done
      // ---- stage weight slice [m][o][cc*32..+31]: 4 x 16B per row ----
      for (int t = tid; t < C_OUT * 4; t += 256) {
        int o = t >> 2, seg = t & 3;
        size_t gi = (((size_t)(m * C_OUT + o)) << 7) + (cc << 5) + (seg << 3);
        async_cp16(whi + gi, &WsHi[o * WSTR + (seg << 3)]);
        async_cp16(wlo + gi, &WsLo[o * WSTR + (seg << 3)]);
      }
      async_wait0();
      __syncthreads();

      // ---- A fragments (weights): 16-bit A 16x32 layout ----
      Frag aHi[2], aLo[2], bHi[2], bLo[2];
#pragma unroll
      for (int i = 0; i < 2; ++i) {
        int o  = 32 * wr + 16 * i + l15;
        int c0 = 16 * half;
        const unsigned short* ph = &WsHi[o * WSTR + c0];
        const unsigned short* pl = &WsLo[o * WSTR + c0];
        aHi[i].q[0] = *(const uint4*)(ph);
        aHi[i].q[1] = *(const uint4*)(ph + 8);
        aLo[i].q[0] = *(const uint4*)(pl);
        aLo[i].q[1] = *(const uint4*)(pl + 8);
      }
      // ---- B fragments (signal): 16-bit B 32x16 layout, tap shift = +m cols ----
#pragma unroll
      for (int j = 0; j < 2; ++j) {
        int col = 32 * wc + 16 * j + l15 + m;
        int c0  = cc * 32 + 16 * half;
        const unsigned short* ph = &XsHi[col * XSTR + c0];
        const unsigned short* pl = &XsLo[col * XSTR + c0];
        bHi[j].q[0] = *(const uint4*)(ph);
        bHi[j].q[1] = *(const uint4*)(ph + 8);
        bLo[j].q[0] = *(const uint4*)(pl);
        bLo[j].q[1] = *(const uint4*)(pl + 8);
      }

      // ---- bf16x3: lo terms first, hi*hi last ----
#pragma unroll
      for (int i = 0; i < 2; ++i)
#pragma unroll
        for (int j = 0; j < 2; ++j) {
          acc[i][j] = __builtin_amdgcn_wmma_f32_16x16x32_bf16(
              false, aLo[i].v, false, bHi[j].v, (short)0, acc[i][j], false, false);
          acc[i][j] = __builtin_amdgcn_wmma_f32_16x16x32_bf16(
              false, aHi[i].v, false, bLo[j].v, (short)0, acc[i][j], false, false);
          acc[i][j] = __builtin_amdgcn_wmma_f32_16x16x32_bf16(
              false, aHi[i].v, false, bHi[j].v, (short)0, acc[i][j], false, false);
        }
    }
  }

  float* ob = out + (size_t)b * C_OUT * NLEN;
#pragma unroll
  for (int i = 0; i < 2; ++i)
#pragma unroll
    for (int j = 0; j < 2; ++j) {
      int n = n_base + 32 * wc + 16 * j + l15;
#pragma unroll
      for (int g = 0; g < 8; ++g) {
        int r = 32 * wr + 16 * i + g + 8 * half;
        ob[(size_t)r * NLEN + n] = acc[i][j][g] + biasS[r];
      }
    }
}

// =====================================================================
// Fallback: round-1 all-in-one kernel (in-kernel split, no workspace).
// =====================================================================
__global__ __launch_bounds__(256) void circorr_wmma_fb(
    const float* __restrict__ x, const float* __restrict__ w,
    const float* __restrict__ bias, float* __restrict__ out) {
  __shared__ __align__(16) unsigned short XsHi[XCOLS * XSTR];
  __shared__ __align__(16) unsigned short XsLo[XCOLS * XSTR];
  __shared__ __align__(16) unsigned short WsHi[C_OUT * WSTR];
  __shared__ __align__(16) unsigned short WsLo[C_OUT * WSTR];
  __shared__ float biasS[C_OUT];

  const int tid  = threadIdx.x;
  const int lane = tid & 31;
  const int wave = tid >> 5;
  const int wr   = wave >> 1;
  const int wc   = wave & 1;
  const int l15  = lane & 15;
  const int half = lane >> 4;

  const int nb     = blockIdx.x;
  const int b      = blockIdx.y;
  const int n_base = nb * NT;

  const float* xb = x + (size_t)b * C_IN * NLEN;
  for (int t = tid; t < C_IN * XCOLS; t += 256) {
    int c = t / XCOLS;
    int j = t - c * XCOLS;
    int g = (n_base + j) & (NLEN - 1);
    float v = xb[c * NLEN + g];
    unsigned short h = bf16_rne(v);
    XsHi[j * XSTR + c] = h;
    XsLo[j * XSTR + c] = bf16_rne(v - bf16_tof(h));
  }
  if (tid < C_OUT) biasS[tid] = bias[tid];

  v8f acc[2][2] = {};

  for (int m = 0; m < KSIZE; ++m) {
    for (int cc = 0; cc < C_IN / 32; ++cc) {
      __syncthreads();
      for (int t = tid; t < C_OUT * 32; t += 256) {
        int o  = t >> 5;
        int cr = t & 31;
        float v = w[o * (C_IN * KSIZE) + (cc * 32 + cr) * KSIZE + m];
        unsigned short h = bf16_rne(v);
        WsHi[o * WSTR + cr] = h;
        WsLo[o * WSTR + cr] = bf16_rne(v - bf16_tof(h));
      }
      __syncthreads();

      Frag aHi[2], aLo[2], bHi[2], bLo[2];
#pragma unroll
      for (int i = 0; i < 2; ++i) {
        int o  = 32 * wr + 16 * i + l15;
        int c0 = 16 * half;
        const unsigned short* ph = &WsHi[o * WSTR + c0];
        const unsigned short* pl = &WsLo[o * WSTR + c0];
        aHi[i].q[0] = *(const uint4*)(ph);
        aHi[i].q[1] = *(const uint4*)(ph + 8);
        aLo[i].q[0] = *(const uint4*)(pl);
        aLo[i].q[1] = *(const uint4*)(pl + 8);
      }
#pragma unroll
      for (int j = 0; j < 2; ++j) {
        int col = 32 * wc + 16 * j + l15 + m;
        int c0  = cc * 32 + 16 * half;
        const unsigned short* ph = &XsHi[col * XSTR + c0];
        const unsigned short* pl = &XsLo[col * XSTR + c0];
        bHi[j].q[0] = *(const uint4*)(ph);
        bHi[j].q[1] = *(const uint4*)(ph + 8);
        bLo[j].q[0] = *(const uint4*)(pl);
        bLo[j].q[1] = *(const uint4*)(pl + 8);
      }

#pragma unroll
      for (int i = 0; i < 2; ++i)
#pragma unroll
        for (int j = 0; j < 2; ++j) {
          acc[i][j] = __builtin_amdgcn_wmma_f32_16x16x32_bf16(
              false, aLo[i].v, false, bHi[j].v, (short)0, acc[i][j], false, false);
          acc[i][j] = __builtin_amdgcn_wmma_f32_16x16x32_bf16(
              false, aHi[i].v, false, bLo[j].v, (short)0, acc[i][j], false, false);
          acc[i][j] = __builtin_amdgcn_wmma_f32_16x16x32_bf16(
              false, aHi[i].v, false, bHi[j].v, (short)0, acc[i][j], false, false);
        }
    }
  }

  float* ob = out + (size_t)b * C_OUT * NLEN;
#pragma unroll
  for (int i = 0; i < 2; ++i)
#pragma unroll
    for (int j = 0; j < 2; ++j) {
      int n = n_base + 32 * wc + 16 * j + l15;
#pragma unroll
      for (int g = 0; g < 8; ++g) {
        int r = 32 * wr + 16 * i + g + 8 * half;
        ob[(size_t)r * NLEN + n] = acc[i][j][g] + biasS[r];
      }
    }
}

extern "C" void kernel_launch(void* const* d_in, const int* in_sizes, int n_in,
                              void* d_out, int out_size, void* d_ws, size_t ws_size,
                              hipStream_t stream) {
  (void)in_sizes; (void)n_in; (void)out_size;
  const float* x    = (const float*)d_in[0];
  const float* k    = (const float*)d_in[1];
  const float* bias = (const float*)d_in[2];
  float* out        = (float*)d_out;
  dim3 grid(NLEN / NT, B_SZ);   // 64 position tiles x 32 batches

  if (ws_size >= WS_NEED && d_ws != nullptr) {
    unsigned short* ws16 = (unsigned short*)d_ws;
    unsigned short* xhi = ws16;
    unsigned short* xlo = ws16 + X_ELEMS;
    unsigned short* whi = ws16 + 2 * X_ELEMS;
    unsigned short* wlo = ws16 + 2 * X_ELEMS + W_ELEMS;
    split_x<<<dim3(NLEN / 64, B_SZ), 256, 0, stream>>>(x, xhi, xlo);
    split_w<<<dim3(((int)W_ELEMS + 255) / 256), 256, 0, stream>>>(k, whi, wlo);
    circorr_wmma_async<<<grid, 256, 0, stream>>>(xhi, xlo, whi, wlo, bias, out);
  } else {
    circorr_wmma_fb<<<grid, 256, 0, stream>>>(x, k, bias, out);
  }
}
